// TokenSparse_70720931495980
// MI455X (gfx1250) — matrix operand order repaired and checked by reading, not compiled, not run on hardware
//
#include <hip/hip_runtime.h>
#include <math.h>

// ---------------------------------------------------------------------------
// Problem constants (from reference): B=32, N=4096, C=512, H=128
// num_keep = ceil(4096*0.6) = 2458 ; tail = 1638
// ---------------------------------------------------------------------------
#define B_DIM     32
#define N_DIM     4096
#define C_DIM     512
#define H_DIM     128
#define NUM_KEEP  2458
#define NUM_TAIL  (N_DIM - NUM_KEEP)   // 1638

typedef __attribute__((ext_vector_type(16))) __bf16 v16bf;
typedef __attribute__((ext_vector_type(8)))  float  v8f;

// ---------------------------------------------------------------------------
// Kernel A: fused  s_pred = sigmoid( gelu(tokens@W1 + b1) @ W2 + b2 )
// One wave handles a 16-row M tile. 8 waves / block -> 128 rows per block.
// W1 (512x128 f32) is staged once per block into LDS, transposed, as bf16
// so B-fragments are 32B-contiguous ds_load_b128 pairs.
// K loop outer (tokens read exactly once), 8 f32 accumulators over H tiles.
// ---------------------------------------------------------------------------
__global__ void __launch_bounds__(256)
k_pred_wmma(const float* __restrict__ tokens,
            const float* __restrict__ W1,
            const float* __restrict__ b1,
            const float* __restrict__ W2,
            const float* __restrict__ b2,
            float* __restrict__ s_pred)
{
    extern __shared__ __bf16 w1t[];            // [H_DIM][C_DIM] transposed, 128 KB
    const int tid = threadIdx.x;

    // Stage W1 (row-major KxH) transposed into LDS as bf16: w1t[n*C + k]
    for (int e = tid; e < C_DIM * H_DIM; e += 256) {
        const int k = e >> 7;                  // e / H_DIM
        const int n = e & (H_DIM - 1);         // e % H_DIM
        w1t[n * C_DIM + k] = (__bf16)W1[e];
    }
    __syncthreads();

    const int wave = tid >> 5;                 // wave32
    const int lane = tid & 31;
    const int m    = lane & 15;                // row-in-tile / column selector
    const int kg   = lane >> 4;                // K-group half (A) / K half (B)

    const long tile    = (long)blockIdx.x * 8 + wave;
    const long rowbase = tile * 16;
    const float* arow  = tokens + (rowbase + m) * (long)C_DIM;

    // C/D accumulators, one 16x16 tile per H group, biased with b1
    v8f acc[8];
#pragma unroll
    for (int j = 0; j < 8; ++j) {
        const float bias = b1[j * 16 + m];
#pragma unroll
        for (int i = 0; i < 8; ++i) acc[j][i] = bias;
    }

    for (int k = 0; k < C_DIM; k += 32) {
        // A fragment (16x32 bf16): lanes 0-15 -> K k..k+15,
        // lanes 16-31 -> K k+16..k+31 (contiguous, matches B)
        const float4 a0 = *(const float4*)(arow + k + kg * 16);
        const float4 a1 = *(const float4*)(arow + k + kg * 16 + 4);
        const float4 a2 = *(const float4*)(arow + k + kg * 16 + 8);
        const float4 a3 = *(const float4*)(arow + k + kg * 16 + 12);
        v16bf afrag;
        afrag[0]  = (__bf16)a0.x; afrag[1]  = (__bf16)a0.y;
        afrag[2]  = (__bf16)a0.z; afrag[3]  = (__bf16)a0.w;
        afrag[4]  = (__bf16)a1.x; afrag[5]  = (__bf16)a1.y;
        afrag[6]  = (__bf16)a1.z; afrag[7]  = (__bf16)a1.w;
        afrag[8]  = (__bf16)a2.x; afrag[9]  = (__bf16)a2.y;
        afrag[10] = (__bf16)a2.z; afrag[11] = (__bf16)a2.w;
        afrag[12] = (__bf16)a3.x; afrag[13] = (__bf16)a3.y;
        afrag[14] = (__bf16)a3.z; afrag[15] = (__bf16)a3.w;

#pragma unroll
        for (int j = 0; j < 8; ++j) {
            // B fragment (32x16): lane half selects K 0-15 vs 16-31, col = m.
            // Contiguous 32B in transposed LDS image.
            const v16bf bfrag =
                *(const v16bf*)(w1t + (j * 16 + m) * C_DIM + k + kg * 16);
            acc[j] = __builtin_amdgcn_wmma_f32_16x16x32_bf16(
                false, afrag, false, bfrag, (short)0, acc[j], false, false);
        }
    }

    // Exact GELU then dot with W2 (per-lane partials, col = j*16+m).
    // C layout: VGPR i holds row i (lanes 0-15) / row 8+i (lanes 16-31).
    float part[8];
#pragma unroll
    for (int i = 0; i < 8; ++i) part[i] = 0.0f;
#pragma unroll
    for (int j = 0; j < 8; ++j) {
        const float w2v = W2[j * 16 + m];
#pragma unroll
        for (int i = 0; i < 8; ++i) {
            const float x = acc[j][i];
            const float g = 0.5f * x * (1.0f + erff(x * 0.70710678118654752f));
            part[i] += g * w2v;
        }
    }
    // Reduce over the 16 lanes of each half (masks < 16 stay in-half)
#pragma unroll
    for (int off = 1; off < 16; off <<= 1) {
#pragma unroll
        for (int i = 0; i < 8; ++i) part[i] += __shfl_xor(part[i], off, 32);
    }
    if (m == 0) {                              // lane 0 -> rows 0..7, lane 16 -> rows 8..15
        const float bb = b2[0];
        const long  rb = rowbase + kg * 8;
#pragma unroll
        for (int i = 0; i < 8; ++i) {
            const float v = part[i] + bb;
            s_pred[rb + i] = 1.0f / (1.0f + expf(-v));
        }
    }
}

// ---------------------------------------------------------------------------
// Kernel B: per-batch min/max normalize the 3 attention rows, blend score.
// score = 0.5*s_pred + 0.25*(nm2 + nm3 + 2*nim)
// ---------------------------------------------------------------------------
__global__ void __launch_bounds__(256)
k_score_mix(const float* __restrict__ sa,
            const float* __restrict__ m2,
            const float* __restrict__ m3,
            const float* __restrict__ sp,
            float* __restrict__ score)
{
    __shared__ float red[6][256];
    const int b = blockIdx.x, tid = threadIdx.x;
    const float* src0 = sa + (long)b * N_DIM;
    const float* src1 = m2 + (long)b * N_DIM;
    const float* src2 = m3 + (long)b * N_DIM;

    float mn0 = 3.4e38f, mn1 = 3.4e38f, mn2 = 3.4e38f;
    float mx0 = -3.4e38f, mx1 = -3.4e38f, mx2 = -3.4e38f;
    for (int n = tid; n < N_DIM; n += 256) {
        float v0 = src0[n], v1 = src1[n], v2 = src2[n];
        mn0 = fminf(mn0, v0); mx0 = fmaxf(mx0, v0);
        mn1 = fminf(mn1, v1); mx1 = fmaxf(mx1, v1);
        mn2 = fminf(mn2, v2); mx2 = fmaxf(mx2, v2);
    }
    red[0][tid] = mn0; red[1][tid] = mn1; red[2][tid] = mn2;
    red[3][tid] = mx0; red[4][tid] = mx1; red[5][tid] = mx2;
    __syncthreads();
    for (int s = 128; s > 0; s >>= 1) {
        if (tid < s) {
            red[0][tid] = fminf(red[0][tid], red[0][tid + s]);
            red[1][tid] = fminf(red[1][tid], red[1][tid + s]);
            red[2][tid] = fminf(red[2][tid], red[2][tid + s]);
            red[3][tid] = fmaxf(red[3][tid], red[3][tid + s]);
            red[4][tid] = fmaxf(red[4][tid], red[4][tid + s]);
            red[5][tid] = fmaxf(red[5][tid], red[5][tid + s]);
        }
        __syncthreads();
    }
    const float a0 = red[0][0], s0 = 1.0f / (red[3][0] - red[0][0] + 1e-8f);
    const float a1 = red[1][0], s1 = 1.0f / (red[4][0] - red[1][0] + 1e-8f);
    const float a2 = red[2][0], s2 = 1.0f / (red[5][0] - red[2][0] + 1e-8f);

    for (int n = tid; n < N_DIM; n += 256) {
        const float nim = (src0[n] - a0) * s0;
        const float nm2 = (src1[n] - a1) * s1;
        const float nm3 = (src2[n] - a2) * s2;
        score[(long)b * N_DIM + n] =
            0.5f * sp[(long)b * N_DIM + n] + 0.25f * (nm2 + nm3 + 2.0f * nim);
    }
}

// ---------------------------------------------------------------------------
// Kernel C: per-batch in-LDS bitonic argsort (descending) of 4096 scores.
// Also emits: order[], score_mask (scatter), and tail softmax weights.
// ---------------------------------------------------------------------------
__global__ void __launch_bounds__(1024)
k_sort(const float* __restrict__ score,
       int* __restrict__ order,
       float* __restrict__ tail_w,
       float* __restrict__ mask_out)
{
    __shared__ float key[N_DIM];
    __shared__ int   idx[N_DIM];
    __shared__ float red[1024];
    const int b = blockIdx.x, tid = threadIdx.x;

    for (int i = tid; i < N_DIM; i += 1024) {
        key[i] = score[(long)b * N_DIM + i];
        idx[i] = i;
    }
    __syncthreads();

    for (int k = 2; k <= N_DIM; k <<= 1) {
        for (int j = k >> 1; j > 0; j >>= 1) {
            for (int i = tid; i < N_DIM; i += 1024) {
                const int ixj = i ^ j;
                if (ixj > i) {
                    const bool desc = ((i & k) == 0);
                    const float a = key[i], c = key[ixj];
                    const bool sw = desc ? (a < c) : (a > c);
                    if (sw) {
                        key[i] = c; key[ixj] = a;
                        const int t = idx[i]; idx[i] = idx[ixj]; idx[ixj] = t;
                    }
                }
            }
            __syncthreads();
        }
    }

    // order + scatter mask
    for (int i = tid; i < N_DIM; i += 1024) {
        order[(long)b * N_DIM + i] = idx[i];
        mask_out[(long)b * N_DIM + idx[i]] = (i < NUM_KEEP) ? 1.0f : 0.0f;
    }

    // tail softmax weights (sorted desc -> max of tail is key[NUM_KEEP])
    const float maxv = key[NUM_KEEP];
    float lsum = 0.0f;
    for (int i = NUM_KEEP + tid; i < N_DIM; i += 1024)
        lsum += expf(key[i] - maxv);
    red[tid] = lsum;
    __syncthreads();
    for (int s = 512; s > 0; s >>= 1) {
        if (tid < s) red[tid] += red[tid + s];
        __syncthreads();
    }
    const float inv_denom = 1.0f / red[0];
    for (int i = NUM_KEEP + tid; i < N_DIM; i += 1024)
        tail_w[(long)b * NUM_TAIL + (i - NUM_KEEP)] =
            expf(key[i] - maxv) * inv_denom;
}

// ---------------------------------------------------------------------------
// Kernel D: gather kept tokens (float4, one 512-float row per block)
// ---------------------------------------------------------------------------
__global__ void __launch_bounds__(128)
k_gather(const float* __restrict__ tokens,
         const int* __restrict__ order,
         float* __restrict__ out_select)
{
    const long blk = blockIdx.x;               // b*NUM_KEEP + i
    const int  b   = (int)(blk / NUM_KEEP);
    const int  i   = (int)(blk % NUM_KEEP);
    const int  src = order[(long)b * N_DIM + i];
    const float4* s = (const float4*)(tokens + ((long)b * N_DIM + src) * C_DIM);
    float4*       d = (float4*)(out_select + ((long)b * NUM_KEEP + i) * C_DIM);
    d[threadIdx.x] = s[threadIdx.x];
}

// ---------------------------------------------------------------------------
// Kernel E: extra_token = sum over the 1638 dropped tokens * softmax weight
// one block per batch; thread owns 2 columns (float2)
// ---------------------------------------------------------------------------
__global__ void __launch_bounds__(256)
k_extra(const float* __restrict__ tokens,
        const int* __restrict__ order,
        const float* __restrict__ tail_w,
        float* __restrict__ out_extra)
{
    const int b = blockIdx.x;
    const int col = threadIdx.x * 2;
    float ax = 0.0f, ay = 0.0f;
    for (int r = 0; r < NUM_TAIL; ++r) {
        const int src = order[(long)b * N_DIM + NUM_KEEP + r];
        const float w = tail_w[(long)b * NUM_TAIL + r];
        const float2 t = *(const float2*)(tokens + ((long)b * N_DIM + src) * C_DIM + col);
        ax = fmaf(t.x, w, ax);
        ay = fmaf(t.y, w, ay);
    }
    float2 o; o.x = ax; o.y = ay;
    *(float2*)(out_extra + (long)b * C_DIM + col) = o;
}

// ---------------------------------------------------------------------------
extern "C" void kernel_launch(void* const* d_in, const int* in_sizes, int n_in,
                              void* d_out, int out_size, void* d_ws, size_t ws_size,
                              hipStream_t stream)
{
    (void)in_sizes; (void)n_in; (void)out_size; (void)ws_size;

    const float* tokens = (const float*)d_in[0];
    const float* sa     = (const float*)d_in[1];
    const float* m2     = (const float*)d_in[2];
    const float* m3     = (const float*)d_in[3];
    const float* W1     = (const float*)d_in[4];
    const float* b1     = (const float*)d_in[5];
    const float* W2     = (const float*)d_in[6];
    const float* b2     = (const float*)d_in[7];

    // workspace layout
    float* s_pred = (float*)d_ws;                           // B*N
    float* score  = s_pred + (long)B_DIM * N_DIM;           // B*N
    int*   order  = (int*)(score + (long)B_DIM * N_DIM);    // B*N
    float* tail_w = (float*)(order + (long)B_DIM * N_DIM);  // B*NUM_TAIL

    // output layout: select_tokens | extra_token | score_mask
    float* out_select = (float*)d_out;
    float* out_extra  = out_select + (long)B_DIM * NUM_KEEP * C_DIM;
    float* out_mask   = out_extra + (long)B_DIM * C_DIM;

    const int lds_bytes = H_DIM * C_DIM * sizeof(__bf16);   // 128 KB
    (void)hipFuncSetAttribute((const void*)k_pred_wmma,
                              hipFuncAttributeMaxDynamicSharedMemorySize,
                              lds_bytes);

    // A: fused GEMM+GELU+GEMV+sigmoid. 131072 rows / (8 waves * 16) = 1024 blocks
    k_pred_wmma<<<(B_DIM * N_DIM) / 128, 256, lds_bytes, stream>>>(
        tokens, W1, b1, W2, b2, s_pred);

    // B: score blending
    k_score_mix<<<B_DIM, 256, 0, stream>>>(sa, m2, m3, s_pred, score);

    // C: bitonic argsort + mask + tail softmax weights
    k_sort<<<B_DIM, 1024, 0, stream>>>(score, order, tail_w, out_mask);

    // D: gather kept tokens
    k_gather<<<B_DIM * NUM_KEEP, 128, 0, stream>>>(tokens, order, out_select);

    // E: weighted sum of dropped tokens
    k_extra<<<B_DIM, 256, 0, stream>>>(tokens, order, tail_w, out_extra);
}
